// MoeLayer_49091476193825
// MI455X (gfx1250) — compile-verified, matrix operand order checked by reading, not run on hardware
//
#include <hip/hip_runtime.h>
#include <hip/hip_bf16.h>
#include <math.h>

#define D_MODEL     1024
#define HIDDEN      4096
#define NUM_CLASSES 1000
#define NUM_EXPERTS 8
#define BATCH       8192
#define MTILE       32
#define NPAD        1024
#define HCHUNK      32

typedef __attribute__((ext_vector_type(16))) __bf16 v16bf;
typedef __attribute__((ext_vector_type(8)))  float  v8f;

// ---------------------------------------------------------------------------
// CDNA5 async global->LDS copies (GLOBAL_LOAD_ASYNC_TO_LDS_B128, ASYNCcnt).
// Async loads complete in order -> counted waits drain exactly the chunk we
// need while later-issued DMA stays in flight.
// ---------------------------------------------------------------------------
__device__ __forceinline__ void async_copy_b128(unsigned lds_addr, const void* gaddr) {
  asm volatile("global_load_async_to_lds_b128 %0, %1, off"
               :: "v"(lds_addr), "v"(gaddr) : "memory");
}
__device__ __forceinline__ void wait_async_le32() {
  asm volatile("s_wait_asynccnt 0x20" ::: "memory");
}
__device__ __forceinline__ void wait_async_le16() {
  asm volatile("s_wait_asynccnt 0x10" ::: "memory");
}
__device__ __forceinline__ unsigned lds_off(const void* p) {
  return (unsigned)(unsigned long long)p;   // LDS aperture: low 32 bits = offset
}

// ---------------------------------------------------------------------------
// 16x32 (rows x K) bf16 WMMA A/B fragment from LDS tile [row][k] row-major.
// ISA 7.12.2: lane L -> row base+(L%16); K 8h..8h+7, 16+8h..16+8h+7 (h=L/16); 2 bf16 per VGPR.
// ---------------------------------------------------------------------------
__device__ __forceinline__ v16bf load_frag(const __bf16* base, int stride,
                                           int base_row, int kbase, int lane) {
  union { v16bf v; unsigned int u[8]; } f;
  const __bf16* p = base + (base_row + (lane & 15)) * stride + kbase + ((lane >> 4) << 3);
  const unsigned int* p32 = (const unsigned int*)p;
#pragma unroll
  for (int j = 0; j < 4; ++j) {
    f.u[j]     = p32[j];
    f.u[4 + j] = p32[j + 8];
  }
  return f.v;
}

__device__ __forceinline__ v8f zero_v8f() {
  v8f z;
#pragma unroll
  for (int i = 0; i < 8; ++i) z[i] = 0.f;
  return z;
}

__device__ __forceinline__ unsigned pack2bf(float a, float b) {
  union { __bf16 h[2]; unsigned u; } p;
  p.h[0] = (__bf16)a; p.h[1] = (__bf16)b;
  return p.u;
}

// ---------------------------------------------------------------------------
__global__ void zero_kernel(float* __restrict__ out, int* __restrict__ counts) {
  size_t i = (size_t)blockIdx.x * 1024u + threadIdx.x;
  if (i < (size_t)BATCH * NUM_CLASSES) out[i] = 0.f;
  if (i < NUM_EXPERTS) counts[i] = 0;
}

// ---------------------------------------------------------------------------
// Tiled transpose + f32->bf16 convert (one-time per call).
// src [R][src_stride] f32 (cols < Cvalid valid) -> dst [Cpad][R] bf16.
// ---------------------------------------------------------------------------
__global__ __launch_bounds__(256)
void transpose_cvt_kernel(const float* __restrict__ src, __bf16* __restrict__ dst,
                          int R, int src_stride, int Cvalid,
                          size_t src_estride, size_t dst_estride) {
  __shared__ __bf16 tile[64][65];
  const float* S = src + (size_t)blockIdx.z * src_estride;
  __bf16*      D = dst + (size_t)blockIdx.z * dst_estride;
  const int c0 = blockIdx.x * 64;
  const int r0 = blockIdx.y * 64;
  const int tid = threadIdx.x;
  const int rr = tid >> 2;
  const int cs = (tid & 3) * 16;
  const float* srow = S + (size_t)(r0 + rr) * src_stride;
#pragma unroll
  for (int j = 0; j < 16; ++j) {
    int c = c0 + cs + j;
    float v = (c < Cvalid) ? srow[c] : 0.f;
    tile[rr][cs + j] = (__bf16)v;
  }
  __syncthreads();
  const int d = tid >> 2;
  unsigned* drow = (unsigned*)(D + (size_t)(c0 + d) * R + r0 + cs);
#pragma unroll
  for (int j = 0; j < 8; ++j) {
    union { __bf16 h[2]; unsigned u; } p;
    p.h[0] = tile[cs + 2 * j + 0][d];
    p.h[1] = tile[cs + 2 * j + 1][d];
    drow[j] = p.u;
  }
}

// ---------------------------------------------------------------------------
// Router: one wave32 per token, shfl reduction, top-2 + 2-way softmax,
// builds per-expert token lists.
// ---------------------------------------------------------------------------
__global__ __launch_bounds__(256)
void router_kernel(const float* __restrict__ x, const float* __restrict__ Wg,
                   const float* __restrict__ bg, float* __restrict__ gates,
                   int* __restrict__ topi, int* __restrict__ counts,
                   int* __restrict__ lists, float* __restrict__ gws) {
  const int wave = threadIdx.x >> 5;
  const int lane = threadIdx.x & 31;
  const int token = blockIdx.x * 8 + wave;
  if (token >= BATCH) return;

  float acc[NUM_EXPERTS];
#pragma unroll
  for (int e = 0; e < NUM_EXPERTS; ++e) acc[e] = 0.f;

  const float* xr = x + (size_t)token * D_MODEL;
  for (int k = lane; k < D_MODEL; k += 32) {
    float xv = xr[k];
    const float* wr = Wg + (size_t)k * NUM_EXPERTS;
#pragma unroll
    for (int e = 0; e < NUM_EXPERTS; ++e) acc[e] += xv * wr[e];
  }
#pragma unroll
  for (int off = 16; off >= 1; off >>= 1) {
#pragma unroll
    for (int e = 0; e < NUM_EXPERTS; ++e) acc[e] += __shfl_down(acc[e], off, 32);
  }
  if (lane == 0) {
#pragma unroll
    for (int e = 0; e < NUM_EXPERTS; ++e) acc[e] += bg[e];
    int i0 = 0;
#pragma unroll
    for (int e = 1; e < NUM_EXPERTS; ++e) if (acc[e] > acc[i0]) i0 = e;
    int i1 = (i0 == 0) ? 1 : 0;
#pragma unroll
    for (int e = 0; e < NUM_EXPERTS; ++e)
      if (e != i0 && acc[e] > acc[i1]) i1 = e;
    float e1 = expf(acc[i1] - acc[i0]);
    float g0 = 1.f / (1.f + e1);
    float g1 = e1 * g0;
#pragma unroll
    for (int e = 0; e < NUM_EXPERTS; ++e) gates[token * NUM_EXPERTS + e] = 0.f;
    gates[token * NUM_EXPERTS + i0] = g0;
    gates[token * NUM_EXPERTS + i1] = g1;
    topi[token * 2 + 0] = i0;
    topi[token * 2 + 1] = i1;
    int p0 = atomicAdd(&counts[i0], 1);
    lists[i0 * BATCH + p0] = token;  gws[i0 * BATCH + p0] = g0;
    int p1 = atomicAdd(&counts[i1], 1);
    lists[i1 * BATCH + p1] = token;  gws[i1 * BATCH + p1] = g1;
  }
}

// ---------------------------------------------------------------------------
// Fused routed expert MLP. Per h-chunk of 32 (software pipelined):
//   W1(i+1) DMA issued at top (hides behind whole iteration),
//   stage 1 on ALL 8 waves (K split 2x512, f32 partials in LDS),
//   partial-sum + bias + GELU + bf16 pack by 256 threads,
//   stage 2: 128 WMMA tiles, accumulators resident in VGPRs,
//   W2(i+1) DMA issued after stage 2 (hides behind next stage 1).
// ---------------------------------------------------------------------------
__global__ __launch_bounds__(256, 1)
void moe_expert_kernel(const float* __restrict__ x,
                       const __bf16* __restrict__ W1T,  // [e][h][k]     bf16
                       const __bf16* __restrict__ W2T,  // [e][n pad][k] bf16
                       const float* __restrict__ b1, const float* __restrict__ b2,
                       const int* __restrict__ counts, const int* __restrict__ lists,
                       const float* __restrict__ gws, float* __restrict__ out) {
  __shared__ __align__(16) __bf16 Xs [MTILE][D_MODEL];      //  64 KB
  __shared__ __align__(16) __bf16 W1s[2][HCHUNK][D_MODEL];  // 128 KB double buf
  __shared__ __align__(16) __bf16 W2s[NPAD][HCHUNK];        //  64 KB
  __shared__ __align__(16) float  Hpart[2][MTILE][HCHUNK];  //   8 KB K-partials
  __shared__ __align__(16) __bf16 Hs [MTILE][HCHUNK];       //   2 KB
  __shared__ int   tok_sh[MTILE];
  __shared__ float gate_sh[MTILE];

  const int e  = blockIdx.x >> 8;
  const int mt = blockIdx.x & 255;
  const int ne = counts[e];
  if (mt * MTILE >= ne) return;   // uniform exit, EXEC stays full for WMMA

  const int tid  = threadIdx.x;
  const int wave = tid >> 5;
  const int lane = tid & 31;

  if (tid < MTILE) {
    int gi = mt * MTILE + tid;
    tok_sh[tid]  = (gi < ne) ? lists[e * BATCH + gi] : 0;
    gate_sh[tid] = (gi < ne) ? gws[e * BATCH + gi] : 0.f;
  }
  __syncthreads();

  // Gather X tile -> LDS bf16 (packed 8B stores).
  {
    int r  = tid >> 3;
    int c0 = (tid & 7) * 128;
    const float4* src = (const float4*)(x + (size_t)tok_sh[r] * D_MODEL + c0);
#pragma unroll
    for (int i = 0; i < 32; ++i) {
      float4 v = src[i];
      uint2 pk; pk.x = pack2bf(v.x, v.y); pk.y = pack2bf(v.z, v.w);
      *(uint2*)&Xs[r][c0 + i * 4] = pk;
    }
  }

  const __bf16* W1Te = W1T + (size_t)e * HIDDEN * D_MODEL;
  const __bf16* W2Te = W2T + (size_t)e * NPAD * HIDDEN;
  const float*  b1e  = b1 + e * HIDDEN;

  const int mi    = wave & 1;     // stage-2 M-half
  const int nbase = wave >> 1;    // stage-2 N-tile base
  v8f acc[16];
#pragma unroll
  for (int t = 0; t < 16; ++t) acc[t] = zero_v8f();

  // stage-1: 8 waves = 4 tiles x 2 K-halves
  const int s1_t  = wave & 3;
  const int s1_kh = wave >> 2;
  const int s1_mi = s1_t & 1;
  const int s1_ni = s1_t >> 1;

  const unsigned w1b[2] = { lds_off(&W1s[0][0][0]), lds_off(&W1s[1][0][0]) };
  const unsigned w2b    = lds_off(&W2s[0][0]);

  // DMA issue helpers (16 b128 ops each per thread-loop => per-wave ASYNCcnt 16)
  auto dma_w1 = [&](int h, unsigned dst) {
    const char* g = (const char*)(W1Te + (size_t)h * D_MODEL);
#pragma unroll
    for (int i = 0; i < 16; ++i) {
      unsigned off = (unsigned)(tid + i * 256) * 16u;
      async_copy_b128(dst + off, g + off);
    }
  };
  auto dma_w2 = [&](int h) {
#pragma unroll
    for (int i = 0; i < 16; ++i) {
      int idx = tid + i * 256;          // 0..4095
      int n = idx >> 2, p4 = idx & 3;
      const char* g = (const char*)(W2Te + (size_t)n * HIDDEN + h) + p4 * 16;
      async_copy_b128(w2b + (unsigned)(n * 64 + p4 * 16), g);
    }
  };

  // Prologue: chunk 0 in flight.
  dma_w1(0, w1b[0]);
  dma_w2(0);

  int cur = 0;
  for (int h0 = 0; h0 < HIDDEN; h0 += HCHUNK) {
    const int hn = (h0 + HCHUNK) & (HIDDEN - 1);   // wrap: last-iter issue is dead but keeps waits uniform
    dma_w1(hn, w1b[cur ^ 1]);       // outstanding: W1(i),W2(i),W1(i+1) <= 48
    wait_async_le32();              // in-order completion => W1(i) resident
    __syncthreads();

    // ---- stage 1: all 8 waves, K-split 2x512, f32 partials ----
    {
      const __bf16* w1buf = &W1s[cur][0][0];
      const int k0 = s1_kh * (D_MODEL / 2);
      v8f h_acc = zero_v8f();
      v16bf a = load_frag(&Xs[0][0], D_MODEL, s1_mi * 16, k0, lane);
      v16bf b = load_frag(w1buf,     D_MODEL, s1_ni * 16, k0, lane);
#pragma unroll 4
      for (int kc = k0; kc < k0 + D_MODEL / 2; kc += 32) {
        int kn = (kc + 32 < k0 + D_MODEL / 2) ? kc + 32 : k0;
        v16bf an = load_frag(&Xs[0][0], D_MODEL, s1_mi * 16, kn, lane);
        v16bf bn = load_frag(w1buf,     D_MODEL, s1_ni * 16, kn, lane);
        h_acc = __builtin_amdgcn_wmma_f32_16x16x32_bf16(
            false, a, false, b, (short)0, h_acc, false, false);
        a = an; b = bn;
      }
      int col = s1_ni * 16 + (lane & 15);
      int rb  = s1_mi * 16 + ((lane >> 4) << 3);
#pragma unroll
      for (int v = 0; v < 8; ++v) Hpart[s1_kh][rb + v][col] = h_acc[v];
    }
    wait_async_le16();              // W2(i) resident (W1(i+1) may stay in flight)
    __syncthreads();

    // ---- partial sum + bias + exact GELU -> Hs (bf16) ----
    {
      int row = tid >> 3;
      int cb  = (tid & 7) * 4;
      uint2 pk;
      __bf16 h4[4];
#pragma unroll
      for (int j = 0; j < 4; ++j) {
        float hv = Hpart[0][row][cb + j] + Hpart[1][row][cb + j] + b1e[h0 + cb + j];
        float g  = 0.5f * hv * (1.f + erff(hv * 0.70710678118654752f));
        h4[j] = (__bf16)g;
      }
      pk.x = *(unsigned*)&h4[0];
      pk.y = *(unsigned*)&h4[2];
      *(uint2*)&Hs[row][cb] = pk;
    }
    __syncthreads();

    // ---- stage 2: OUT[32x1024] += Hs @ W2chunk, K=32, 16 tiles/wave ----
    {
      v16bf a2 = load_frag(&Hs[0][0], HCHUNK, mi * 16, 0, lane);
      v16bf bf = load_frag(&W2s[0][0], HCHUNK, nbase * 16, 0, lane);
#pragma unroll
      for (int t = 0; t < 16; ++t) {
        int tn = (t < 15) ? t + 1 : 0;
        v16bf bn = load_frag(&W2s[0][0], HCHUNK, (nbase + 4 * tn) * 16, 0, lane);
        acc[t] = __builtin_amdgcn_wmma_f32_16x16x32_bf16(
            false, a2, false, bf, (short)0, acc[t], false, false);
        bf = bn;
      }
    }
    __syncthreads();                // all waves done reading W2s
    dma_w2(hn);                     // W2(i+1): hides behind next stage 1
    cur ^= 1;
  }

  // ---- scatter: out[token] += gate * (acc + b2) ----
  const int colL  = lane & 15;
  const int rbase = mi * 16 + ((lane >> 4) << 3);
#pragma unroll
  for (int t = 0; t < 16; ++t) {
    int col = (nbase + 4 * t) * 16 + colL;
    if (col >= NUM_CLASSES) continue;
    float b2v = b2[e * NUM_CLASSES + col];
#pragma unroll
    for (int v = 0; v < 8; ++v) {
      int row = rbase + v;
      if (mt * MTILE + row < ne) {
        atomicAdd(&out[(size_t)tok_sh[row] * NUM_CLASSES + col],
                  gate_sh[row] * (acc[t][v] + b2v));
      }
    }
  }
}

// ---------------------------------------------------------------------------
extern "C" void kernel_launch(void* const* d_in, const int* in_sizes, int n_in,
                              void* d_out, int out_size, void* d_ws, size_t ws_size,
                              hipStream_t stream) {
  const float* x  = (const float*)d_in[0];
  const float* Wg = (const float*)d_in[1];
  const float* bg = (const float*)d_in[2];
  const float* W1 = (const float*)d_in[3];
  const float* b1 = (const float*)d_in[4];
  const float* W2 = (const float*)d_in[5];
  const float* b2 = (const float*)d_in[6];

  float* out   = (float*)d_out;                               // [8192,1000]
  float* gates = out + (size_t)BATCH * NUM_CLASSES;           // [8192,8]
  int*   topi  = (int*)(gates + (size_t)BATCH * NUM_EXPERTS); // [8192,2]

  // ws: counts[8] | lists int[8][8192] | gws float[8][8192] | W1T 64MB | W2T 64MB
  char* ws = (char*)d_ws;
  int*    counts = (int*)ws;
  int*    lists  = (int*)(ws + 256);
  float*  gws    = (float*)(ws + 256 + sizeof(int) * NUM_EXPERTS * BATCH);
  __bf16* W1T    = (__bf16*)(ws + (1ull << 20));
  __bf16* W2T    = (__bf16*)(ws + (1ull << 20) + ((size_t)NUM_EXPERTS * HIDDEN * D_MODEL * 2));

  zero_kernel<<<(BATCH * NUM_CLASSES + 1023) / 1024, 1024, 0, stream>>>(out, counts);
  router_kernel<<<BATCH / 8, 256, 0, stream>>>(x, Wg, bg, gates, topi,
                                               counts, lists, gws);
  // W1 [1024 k][4096 h] -> W1T [4096 h][1024 k]
  transpose_cvt_kernel<<<dim3(HIDDEN / 64, D_MODEL / 64, NUM_EXPERTS), 256, 0, stream>>>(
      W1, W1T, D_MODEL, HIDDEN, HIDDEN,
      (size_t)D_MODEL * HIDDEN, (size_t)HIDDEN * D_MODEL);
  // W2 [4096 k][1000 n] -> W2T [1024 n(pad)][4096 k]
  transpose_cvt_kernel<<<dim3(NPAD / 64, HIDDEN / 64, NUM_EXPERTS), 256, 0, stream>>>(
      W2, W2T, HIDDEN, NUM_CLASSES, NUM_CLASSES,
      (size_t)HIDDEN * NUM_CLASSES, (size_t)NPAD * HIDDEN);
  moe_expert_kernel<<<NUM_EXPERTS * 256, 256, 0, stream>>>(
      x, W1T, W2T, b1, b2, counts, lists, gws, out);
}
